// SpatialAttention_61710090109621
// MI455X (gfx1250) — matrix operand order repaired and checked by reading, not compiled, not run on hardware
//
#include <hip/hip_runtime.h>
#include <hip/hip_bf16.h>
#include <stdint.h>

typedef __attribute__((ext_vector_type(16))) _Float16     v16h;
typedef __attribute__((ext_vector_type(8)))  float        v8f;
typedef __attribute__((ext_vector_type(8)))  _Float16     h8;
typedef __attribute__((ext_vector_type(4)))  float        f4;
typedef __attribute__((ext_vector_type(4)))  unsigned int u32x4;
typedef __attribute__((ext_vector_type(8)))  int          i32x8;
typedef __attribute__((ext_vector_type(4)))  int          i32x4;

#define TOK   31200     // B*T*N
#define NTOK  325
#define NPAD  352       // 22*16, padded spatial dim
#define BT    96
#define DOUT  64
#define BN_EPS 1e-5f

#define CAT16(lo, hi) __builtin_shufflevector(lo, hi, 0, 1, 2, 3, 4, 5, 6, 7, \
                                              8, 9, 10, 11, 12, 13, 14, 15)

#if defined(__has_builtin)
#if __has_builtin(__builtin_amdgcn_tensor_load_to_lds) && __has_builtin(__builtin_amdgcn_s_wait_tensorcnt)
#define HAVE_TDM 1
#endif
#endif
#ifndef HAVE_TDM
#define HAVE_TDM 0
#endif

__device__ __forceinline__ float bn_act(float y, float sc, float sh) {
    float v = y * sc + sh;
    return v > 0.f ? v : 0.f;
}

#if HAVE_TDM
// 1-D contiguous TDM copy: n_elem 2-byte elements, global -> LDS.
// D# per CDNA5 ISA (count=1, type=2, data_size=2B,
// tensor_dim0 = tile_dim0 = stride = n_elem, higher dims trivial).
__device__ __forceinline__ void tdm_load_1d(const void* gsrc, void* lds_dst,
                                            unsigned int n_elem) {
    uint64_t ga = (uint64_t)(uintptr_t)gsrc;
    unsigned int lds = (unsigned int)(uintptr_t)lds_dst; // low 32 bits = LDS offset
    u32x4 g0;
    g0[0] = 1u;                                               // count = 1
    g0[1] = lds;                                              // lds_addr
    g0[2] = (unsigned int)ga;                                 // global_addr lo
    g0[3] = ((unsigned int)(ga >> 32) & 0x01FFFFFFu) | (2u << 30); // hi | type=2
    i32x8 g1;
    g1[0] = (int)(1u << 16);                                  // data_size = 2B, mask=0
    g1[1] = (int)((n_elem & 0xFFFFu) << 16);                  // tensor_dim0 lo16
    g1[2] = (int)(((n_elem >> 16) & 0xFFFFu) | (1u << 16));   // dim0 hi | tensor_dim1=1
    g1[3] = (int)((n_elem & 0xFFFFu) << 16);                  // tile_dim0
    g1[4] = 1;                                                // tile_dim1 = 1
    g1[5] = (int)n_elem;                                      // tensor_dim0_stride lo
    g1[6] = 0;
    g1[7] = 0;
    i32x4 z = {0, 0, 0, 0};
#if defined(__clang_major__) && __clang_major__ >= 23
    i32x8 z8 = {0, 0, 0, 0, 0, 0, 0, 0};
    __builtin_amdgcn_tensor_load_to_lds(g0, g1, z, z, z8, 0);
#else
    __builtin_amdgcn_tensor_load_to_lds(g0, g1, z, z, 0);
#endif
}
#endif

// ---------------------------------------------------------------------------
__global__ void zero_kernel(float* __restrict__ p, int n) {
    int i = blockIdx.x * blockDim.x + threadIdx.x;
    if (i < n) p[i] = 0.f;
}

// ---------------------------------------------------------------------------
// q/k/v projection. Block = 128 threads (4 waves = 4 channel tiles), one
// 16-token tile, loop over the 3 matrices reusing the staged x-tile.
// ---------------------------------------------------------------------------
__global__ __launch_bounds__(128)
void proj_qkv_kernel(const float* __restrict__ X,  const float* __restrict__ SE,
                     const float* __restrict__ Wq, const float* __restrict__ bq,
                     const float* __restrict__ Wk, const float* __restrict__ bk,
                     const float* __restrict__ Wv, const float* __restrict__ bv,
                     float* __restrict__ Y,            // [3][TOK][64]
                     float* __restrict__ ssum, float* __restrict__ ssqs)
{
    __shared__ _Float16 Xs[16][144];   // 128 used, stride 288B (16B mult)
    __shared__ _Float16 Wt[64][144];   // W^T

    const int tile = blockIdx.x;
    const int tid  = threadIdx.x;
    const int wave = tid >> 5;
    const int lane = tid & 31;
    const int lm   = lane & 15;
    const int hi   = (lane >= 16) ? 1 : 0;
    const int ch   = wave * 16 + lm;

    for (int i = tid; i < 16 * 128; i += 128) {
        int r = i >> 7, c0 = i & 127;
        float xv = (c0 < 64) ? X[((size_t)tile * 16 + r) * 64 + c0]
                             : SE[((size_t)tile * 16 + r) * 64 + (c0 - 64)];
        Xs[r][c0] = (_Float16)xv;
    }

    for (int mat = 0; mat < 3; ++mat) {
        const float* W    = (mat == 0) ? Wq : (mat == 1) ? Wk : Wv;
        const float* bias = (mat == 0) ? bq : (mat == 1) ? bk : bv;
        __syncthreads();                       // Xs ready / prev Wt consumed
        for (int e = tid; e < 128 * 64; e += 128) {
            int r = e >> 6, c0 = e & 63;
            Wt[c0][r] = (_Float16)W[e];        // transpose: Wt[out][in]
        }
        __syncthreads();

        v8f c = {};
        for (int kc = 0; kc < 4; ++kc) {
            h8 a0 = *(const h8*)&Xs[lm][kc * 32 + hi * 8];
            h8 a1 = *(const h8*)&Xs[lm][kc * 32 + hi * 8 + 16];
            h8 b0 = *(const h8*)&Wt[ch][kc * 32 + hi * 8];
            h8 b1 = *(const h8*)&Wt[ch][kc * 32 + hi * 8 + 16];
            v16h a = CAT16(a0, a1);
            v16h b = CAT16(b0, b1);
            c = __builtin_amdgcn_wmma_f32_16x16x32_f16(false, a, false, b,
                                                       (short)0, c, false, false);
        }

        float bch = bias[ch], s = 0.f, s2 = 0.f;
#pragma unroll
        for (int r = 0; r < 8; ++r) {
            float y = c[r] + bch;
            Y[((size_t)mat * TOK + (size_t)tile * 16 + r + hi * 8) * DOUT + ch] = y;
            s += y; s2 += y * y;
        }
        atomicAdd(&ssum[mat * DOUT + ch], s);
        atomicAdd(&ssqs[mat * DOUT + ch], s2);
    }
}

// ---------------------------------------------------------------------------
__global__ void finalize_kernel(const float* __restrict__ ssum,
                                const float* __restrict__ ssqs,
                                const float* __restrict__ gamma,
                                const float* __restrict__ beta,
                                float* __restrict__ scale,
                                float* __restrict__ shift)
{
    int ch = threadIdx.x;   // 64
    const float inv_n = 1.f / (float)TOK;
    float mu  = ssum[ch] * inv_n;
    float var = ssqs[ch] * inv_n - mu * mu;
    float sc  = gamma[ch] * rsqrtf(var + BN_EPS);
    scale[ch] = sc;
    shift[ch] = beta[ch] - mu * sc;
}

// ---------------------------------------------------------------------------
// BN+ReLU applied once; outputs f16 head-major padded layouts:
//   qn,kn : [bt*8+h][352][8]  (rows >= 325 zero; 1/sqrt(8) folded into q)
//   vt    : [bt*8+h][8][352]  (transposed)
// ---------------------------------------------------------------------------
__global__ void normalize_kernel(const float* __restrict__ Yq,
                                 const float* __restrict__ Yk,
                                 const float* __restrict__ Yv,
                                 const float* __restrict__ sscl,
                                 const float* __restrict__ ssft,
                                 _Float16* __restrict__ qn,
                                 _Float16* __restrict__ kn,
                                 _Float16* __restrict__ vt)
{
    const int tid = threadIdx.x;
    const int d   = tid & 7;
    const int nn  = blockIdx.x * 32 + (tid >> 3);   // 0..351
    const int bth = blockIdx.y;
    const int bt  = bth >> 3, h = bth & 7;
    const int ch  = h * 8 + d;

    float q = 0.f, k = 0.f, v = 0.f;
    if (nn < NTOK) {
        size_t src = ((size_t)bt * NTOK + nn) * DOUT + ch;
        q = bn_act(Yq[src], sscl[ch],       ssft[ch])       * 0.35355339f;
        k = bn_act(Yk[src], sscl[64 + ch],  ssft[64 + ch]);
        v = bn_act(Yv[src], sscl[128 + ch], ssft[128 + ch]);
    }
    size_t strip = (size_t)bth * (NPAD * 8);
    qn[strip + (size_t)nn * 8 + d]    = (_Float16)q;
    kn[strip + (size_t)nn * 8 + d]    = (_Float16)k;
    vt[strip + (size_t)d * NPAD + nn] = (_Float16)v;
}

// ---------------------------------------------------------------------------
// Fused attention. Block = 64 threads (2 waves); wave w owns row tile
// ntile = bx*2 + w. k/v strips arrive in LDS via TDM. Since qa is strictly
// zero in all fake-K slots, kb needs no masking; vb lanes d>=8 are garbage
// but their stores are masked. Softmax: 32 lanes, f4-vectorized, halves
// combined with shfl_xor(16).
// ---------------------------------------------------------------------------
__global__ __launch_bounds__(64)
void attn_kernel(const _Float16* __restrict__ qn, const _Float16* __restrict__ kn,
                 const _Float16* __restrict__ vt, float* __restrict__ O)
{
    __shared__ _Float16 Ks[NPAD * 8];     // [m][d]
    __shared__ _Float16 Vs[8 * NPAD];     // [d][m]
    __shared__ float    S[2][16][356];    // per-wave score strip
    __shared__ float    rowinv[2][16];

    const int tid  = threadIdx.x;
    const int wave = tid >> 5;
    const int lane = tid & 31;
    const int lm   = lane & 15;
    const int hi   = (lane >= 16) ? 1 : 0;
    const int bth  = blockIdx.y;
    const int ntile = blockIdx.x * 2 + wave;          // 0..21 (21: stores masked)
    const size_t strip = (size_t)bth * (NPAD * 8);

#if HAVE_TDM
    if (wave == 0) {
        tdm_load_1d(kn + strip, Ks, NPAD * 8);
        tdm_load_1d(vt + strip, Vs, NPAD * 8);
    }
#else
    for (int i = tid; i < NPAD; i += 64) {
        *(h8*)(Ks + i * 8) = *(const h8*)(kn + strip + (size_t)i * 8);
        *(h8*)(Vs + i * 8) = *(const h8*)(vt + strip + (size_t)i * 8);
    }
#endif

    // q fragment (global, overlaps the strip transfer); zero in fake-K slots
    v16h qa;
    {
        const int qrow = ntile * 16 + lm;             // < 352 always
        h8 qr = *(const h8*)(qn + strip + (size_t)qrow * 8);
#pragma unroll
        for (int j = 0; j < 16; ++j)
            qa[j] = (!hi && j < 8) ? qr[j] : (_Float16)0.f;
    }

#if HAVE_TDM
    if (wave == 0) __builtin_amdgcn_s_wait_tensorcnt(0);
#endif
    __syncthreads();

    // ---- scores s[n,m] = q . k (fake-K of kb irrelevant: qa zero there) ----
    for (int mt = 0; mt < 21; ++mt) {
        const int m = mt * 16 + lm;
        h8 kr = *(const h8*)(Ks + m * 8);
        v16h kb = CAT16(kr, kr);
        v8f s = {};
        s = __builtin_amdgcn_wmma_f32_16x16x32_f16(false, qa, false, kb,
                                                   (short)0, s, false, false);
#pragma unroll
        for (int r = 0; r < 8; ++r)
            S[wave][r + hi * 8][m] = s[r];
    }
    // mask invalid columns 325..351 (16 rows x 27 cols)
    for (int p = lane; p < 16 * 27; p += 32)
        S[wave][p & 15][NTOK + (p >> 4)] = -1e30f;
    __syncthreads();

    // ---- softmax over m: 32 lanes, row = lm, half = hi (176 cols each) ----
    {
        float* rowp = &S[wave][lm][0] + hi * 176;
        float mx = -1e30f;
        for (int m = 0; m < 176; m += 4) {
            f4 v = *(const f4*)(rowp + m);
            mx = fmaxf(fmaxf(mx, fmaxf(v[0], v[1])), fmaxf(v[2], v[3]));
        }
        mx = fmaxf(mx, __shfl_xor(mx, 16, 32));
        float sum = 0.f;
        for (int m = 0; m < 176; m += 4) {
            f4 v = *(const f4*)(rowp + m);
            f4 p;
            p[0] = __expf(v[0] - mx);
            p[1] = __expf(v[1] - mx);
            p[2] = __expf(v[2] - mx);
            p[3] = __expf(v[3] - mx);
            *(f4*)(rowp + m) = p;
            sum += (p[0] + p[1]) + (p[2] + p[3]);
        }
        sum += __shfl_xor(sum, 16, 32);
        if (!hi) rowinv[wave][lm] = 1.f / sum;
    }
    __syncthreads();

    // ---- out = P @ V : K = 352 -> 11 WMMA steps ----
    v8f c = {};
    const float* srow = &S[wave][lm][0];
    const _Float16* vrow = Vs + (lm & 7) * NPAD;
    for (int kc = 0; kc < 11; ++kc) {
        f4 p0 = *(const f4*)(srow + kc * 32 + hi * 8);
        f4 p1 = *(const f4*)(srow + kc * 32 + hi * 8 + 4);
        f4 p2 = *(const f4*)(srow + kc * 32 + hi * 8 + 16);
        f4 p3 = *(const f4*)(srow + kc * 32 + hi * 8 + 20);
        h8 v0 = *(const h8*)(vrow + kc * 32 + hi * 8);
        h8 v1 = *(const h8*)(vrow + kc * 32 + hi * 8 + 16);
        v16h pa;
#pragma unroll
        for (int j = 0; j < 4; ++j) {
            pa[j]      = (_Float16)p0[j];
            pa[j + 4]  = (_Float16)p1[j];
            pa[j + 8]  = (_Float16)p2[j];
            pa[j + 12] = (_Float16)p3[j];
        }
        v16h vb = CAT16(v0, v1);
        c = __builtin_amdgcn_wmma_f32_16x16x32_f16(false, pa, false, vb,
                                                   (short)0, c, false, false);
    }

    if (lm < 8) {
        const int h = bth & 7;
        const size_t base = (size_t)(bth >> 3) * NTOK;
#pragma unroll
        for (int r = 0; r < 8; ++r) {
            int rr  = r + hi * 8;
            int row = ntile * 16 + rr;
            if (row < NTOK)
                O[(base + row) * DOUT + h * 8 + lm] = c[r] * rowinv[wave][rr];
        }
    }
}

// ---------------------------------------------------------------------------
// Output projection (K=64 -> 2 WMMA steps) + o stats. Block = 128 (4 waves).
// ---------------------------------------------------------------------------
__global__ __launch_bounds__(128)
void proj_o_kernel(const float* __restrict__ O,  const float* __restrict__ Wo,
                   const float* __restrict__ bo, float* __restrict__ Yo,
                   float* __restrict__ ssum, float* __restrict__ ssqs)
{
    __shared__ _Float16 Os[16][72];
    __shared__ _Float16 Wt[64][72];

    const int tile = blockIdx.x;
    const int tid  = threadIdx.x;
    const int wave = tid >> 5;
    const int lane = tid & 31;
    const int lm   = lane & 15;
    const int hi   = (lane >= 16) ? 1 : 0;
    const int ch   = wave * 16 + lm;

    for (int i = tid; i < 16 * 64; i += 128) {
        int r = i >> 6, c0 = i & 63;
        Os[r][c0] = (_Float16)O[((size_t)tile * 16 + r) * 64 + c0];
    }
    for (int e = tid; e < 64 * 64; e += 128) {
        int r = e >> 6, c0 = e & 63;
        Wt[c0][r] = (_Float16)Wo[e];
    }
    __syncthreads();

    v8f c = {};
    for (int kc = 0; kc < 2; ++kc) {
        h8 a0 = *(const h8*)&Os[lm][kc * 32 + hi * 8];
        h8 a1 = *(const h8*)&Os[lm][kc * 32 + hi * 8 + 16];
        h8 b0 = *(const h8*)&Wt[ch][kc * 32 + hi * 8];
        h8 b1 = *(const h8*)&Wt[ch][kc * 32 + hi * 8 + 16];
        v16h a = CAT16(a0, a1);
        v16h b = CAT16(b0, b1);
        c = __builtin_amdgcn_wmma_f32_16x16x32_f16(false, a, false, b,
                                                   (short)0, c, false, false);
    }

    float bch = bo[ch], s = 0.f, s2 = 0.f;
#pragma unroll
    for (int r = 0; r < 8; ++r) {
        float y = c[r] + bch;
        Yo[((size_t)tile * 16 + r + hi * 8) * DOUT + ch] = y;
        s += y; s2 += y * y;
    }
    atomicAdd(&ssum[3 * DOUT + ch], s);
    atomicAdd(&ssqs[3 * DOUT + ch], s2);
}

// ---------------------------------------------------------------------------
__global__ void bn_out_kernel(const float* __restrict__ Yo,
                              const float* __restrict__ sc,
                              const float* __restrict__ sh,
                              float* __restrict__ out, int ntot)
{
    int i = blockIdx.x * blockDim.x + threadIdx.x;
    if (i < ntot) {
        int ch = i & 63;
        out[i] = bn_act(Yo[i], sc[ch], sh[ch]);
    }
}

// ---------------------------------------------------------------------------
extern "C" void kernel_launch(void* const* d_in, const int* in_sizes, int n_in,
                              void* d_out, int out_size, void* d_ws, size_t ws_size,
                              hipStream_t stream)
{
    const float* X   = (const float*)d_in[0];
    const float* SE  = (const float*)d_in[1];
    const float* Wq  = (const float*)d_in[2];
    const float* bq  = (const float*)d_in[3];
    const float* gq  = (const float*)d_in[4];
    const float* beq = (const float*)d_in[5];
    const float* Wk  = (const float*)d_in[6];
    const float* bk  = (const float*)d_in[7];
    const float* gk  = (const float*)d_in[8];
    const float* bek = (const float*)d_in[9];
    const float* Wv  = (const float*)d_in[10];
    const float* bv  = (const float*)d_in[11];
    const float* gv  = (const float*)d_in[12];
    const float* bev = (const float*)d_in[13];
    const float* Wo  = (const float*)d_in[14];
    const float* bo  = (const float*)d_in[15];
    const float* go  = (const float*)d_in[16];
    const float* beo = (const float*)d_in[17];
    float* out = (float*)d_out;

    const size_t TOK64 = (size_t)TOK * DOUT;
    const size_t STRIP = (size_t)BT * 8 * NPAD * 8;   // f16 elements per tensor

    float* Yq   = (float*)d_ws;
    float* Yk   = Yq   + TOK64;
    float* Yv   = Yk   + TOK64;
    float* Obuf = Yv   + TOK64;
    float* Yo   = Obuf + TOK64;
    float* ssum = Yo   + TOK64;        // [4][64]
    float* ssqs = ssum + 256;
    float* sscl = ssqs + 256;
    float* ssft = sscl + 256;
    _Float16* qn = (_Float16*)(ssft + 256);
    _Float16* kn = qn + STRIP;
    _Float16* vt = kn + STRIP;

    zero_kernel<<<2, 256, 0, stream>>>(ssum, 512);

    proj_qkv_kernel<<<dim3(TOK / 16), 128, 0, stream>>>(
        X, SE, Wq, bq, Wk, bk, Wv, bv, Yq, ssum, ssqs);

    finalize_kernel<<<1, 64, 0, stream>>>(ssum + 0,   ssqs + 0,   gq, beq, sscl + 0,   ssft + 0);
    finalize_kernel<<<1, 64, 0, stream>>>(ssum + 64,  ssqs + 64,  gk, bek, sscl + 64,  ssft + 64);
    finalize_kernel<<<1, 64, 0, stream>>>(ssum + 128, ssqs + 128, gv, bev, sscl + 128, ssft + 128);

    normalize_kernel<<<dim3(11, BT * 8), 256, 0, stream>>>(
        Yq, Yk, Yv, sscl, ssft, qn, kn, vt);

    attn_kernel<<<dim3(11, BT * 8), 64, 0, stream>>>(qn, kn, vt, Obuf);

    proj_o_kernel<<<dim3(TOK / 16), 128, 0, stream>>>(Obuf, Wo, bo, Yo, ssum, ssqs);

    finalize_kernel<<<1, 64, 0, stream>>>(ssum + 192, ssqs + 192, go, beo, sscl + 192, ssft + 192);

    bn_out_kernel<<<(TOK * DOUT) / 256, 256, 0, stream>>>(Yo, sscl + 192, ssft + 192,
                                                          out, TOK * DOUT);
}
